// StatePredictionNetwork_85847806312938
// MI455X (gfx1250) — compile-verified, matrix operand order checked by reading, not run on hardware
//
#include <hip/hip_runtime.h>
#include <hip/hip_fp16.h>

// ---------------------------------------------------------------------------
// CDNA5 (gfx1250) fused MLP for StatePredictionNetwork
//   B=32, P=128, N=32  ->  131072 rows
//   x(201 pad 224) -> 256 -> 256 -> 128 -> {contrib64, edge64, u128 -> mean/var64}
// f16 WMMA 16x16x32 (f32 accum). Weights pre-packed into B-fragment order in
// d_ws. One block per (b,p) pair (32 rows). Each wave owns a COLUMN tile and
// computes both 16-row tiles, so every weight B-fragment is loaded once and
// feeds two v_wmma ops (2x arithmetic intensity on the weight stream).
// ---------------------------------------------------------------------------

typedef __attribute__((ext_vector_type(16))) _Float16 v16h;
typedef __attribute__((ext_vector_type(8)))  float    v8f;
typedef __attribute__((ext_vector_type(4)))  float    v4f;

#define NWAVES 8

// ---- geometry -------------------------------------------------------------
#define NB   32
#define NP   128
#define NN   32
#define NROW (NB*NP*NN)          // 131072
#define NBP  (NB*NP)             // 4096 blocks

// workspace offsets (in _Float16 elements) for packed weight fragments
#define OFF_W1 0                 // 224x256
#define OFF_W2 57344             // 256x256
#define OFF_W3 122880            // 256x128
#define OFF_WA 155648            // 128x64
#define OFF_WB 163840            // 128x64
#define OFF_WU 172032            // 128x128
#define OFF_WM 188416            // 128x64
#define OFF_WV 196608            // 128x64

// output slabs (floats)
#define OUT_LOG  0
#define OUT_MEAN (NROW)                  // 131072
#define OUT_VAR  (NROW + NROW*64)        // 8519680

// ---------------------------------------------------------------------------
// Weight packing: fragment order matches the CDNA5 16-bit 32x16 B operand.
// Fragment f = (kt*CT + ct): 512 halves, index = lane*16 + e,
//   n = ct*16 + (lane&15)
//   k = kt*32 + (e>>3)*16 + (e&7) + ((lane>>4)<<3)
// ---------------------------------------------------------------------------
__global__ __launch_bounds__(256) void pack_weights_kernel(
    const float* __restrict__ src, _Float16* __restrict__ dst, int K, int NC) {
  const int CT   = NC >> 4;
  const int frag = blockIdx.x;
  const int kt   = frag / CT;
  const int ct   = frag % CT;
  for (int i = threadIdx.x; i < 512; i += 256) {
    const int lane = i >> 4;
    const int e    = i & 15;
    const int n    = ct * 16 + (lane & 15);
    const int k    = kt * 32 + ((e >> 3) << 4) + (e & 7) + ((lane >> 4) << 3);
    const float v  = (k < K) ? src[k * NC + n] : 0.0f;
    dst[frag * 512 + i] = (_Float16)v;
  }
}

// ---------------------------------------------------------------------------
// A-fragment load from row-major LDS (row stride = ldk halves).
// Per lane: m = lane&15; halves e=0..7 cover k=koff..koff+7 (contiguous),
// e=8..15 cover k=koff+16..koff+23 -> two 16-byte LDS loads.
// ---------------------------------------------------------------------------
__device__ __forceinline__ v16h load_a_frag(const _Float16* base, int ldk,
                                            int kt, int lane) {
  const int m    = lane & 15;
  const int koff = kt * 32 + ((lane >> 4) << 3);
  const _Float16* pp = base + m * ldk + koff;
  v16h a;
  ((v4f*)&a)[0] = *(const v4f*)(pp);
  ((v4f*)&a)[1] = *(const v4f*)(pp + 16);
  return a;
}

__device__ __forceinline__ v8f wmma_step(v16h a, v16h b, v8f c) {
  return __builtin_amdgcn_wmma_f32_16x16x32_f16(
      /*neg_a=*/false, a, /*neg_b=*/false, b,
      /*c_mod=*/(short)0, c, /*reuse_a=*/false, /*reuse_b=*/false);
}

__device__ __forceinline__ float log_sigmoid(float x) {
  return fminf(x, 0.0f) - log1pf(expf(-fabsf(x)));
}
__device__ __forceinline__ float softplus_f(float x) {
  return fmaxf(x, 0.0f) + log1pf(expf(-fabsf(x)));
}

// ---------------------------------------------------------------------------
// Column-tile GEMM stage: out[32 x NC] = act(in[32 x KP] @ Wfrag + bias).
// Each wave owns column tiles (ct); for each ct it loads one B fragment per
// k-step and issues two WMMAs (row tiles 0 and 1) -> B loaded once per block.
// ---------------------------------------------------------------------------
template <int KP, int NC, bool RELU>
__device__ __forceinline__ void gemm_f16_lds(
    const _Float16* in, _Float16* out, const _Float16* __restrict__ wfrag,
    const float* __restrict__ bias, int wid, int lane) {
  constexpr int CT = NC / 16;
  constexpr int KT = KP / 32;
  for (int ct = wid; ct < CT; ct += NWAVES) {
    v8f c0 = {}, c1 = {};
#pragma unroll
    for (int kt = 0; kt < KT; ++kt) {
      if (kt + 1 < KT)
        __builtin_prefetch(wfrag + ((kt + 1) * CT + ct) * 512 + lane * 16, 0, 3);
      v16h b  = *(const v16h*)(wfrag + (kt * CT + ct) * 512 + lane * 16);
      v16h a0 = load_a_frag(in,           KP, kt, lane);
      v16h a1 = load_a_frag(in + 16 * KP, KP, kt, lane);
      c0 = wmma_step(a0, b, c0);
      c1 = wmma_step(a1, b, c1);
    }
    const int n  = ct * 16 + (lane & 15);
    const int m0 = (lane >> 4) << 3;
    const float bn = bias[n];
#pragma unroll
    for (int v = 0; v < 8; ++v) {
      float v0 = c0[v] + bn;
      float v1 = c1[v] + bn;
      if (RELU) { v0 = fmaxf(v0, 0.0f); v1 = fmaxf(v1, 0.0f); }
      out[(m0 + v) * NC + n]        = (_Float16)v0;
      out[(16 + m0 + v) * NC + n]   = (_Float16)v1;
    }
  }
}

// ---------------------------------------------------------------------------
struct Params {
  const float *theta, *s, *prev_flag, *prev_acts, *sampled_i;
  const float *b1, *b2, *b3, *be, *Wl, *bl, *bu, *bm, *bv, *wc;
  const _Float16 *w1, *w2, *w3, *wa, *wb, *wu, *wm, *wv;
  float *out;
};

__global__ __launch_bounds__(256) void spn_fused_kernel(Params p) {
  __shared__ __attribute__((aligned(32))) _Float16 Xl[32 * 224];  // 14336 B
  __shared__ __attribute__((aligned(32))) _Float16 Ha[32 * 256];  // 16384 B
  __shared__ __attribute__((aligned(32))) _Float16 Hb[32 * 256];  // 16384 B
  __shared__ float contribF[32 * 64];                             //  8192 B
  __shared__ float edgeF[32 * 64];                                //  8192 B

  const int bp   = blockIdx.x;
  const int R0   = bp * NN;
  const int tid  = threadIdx.x;
  const int wid  = tid >> 5;
  const int lane = tid & 31;

  // ---- stage X = [theta | s | prev_acts | prev_flag | 0-pad] -> f16 LDS ---
  // power-of-two sub-loops: global reads are fully contiguous per section
  for (int idx = tid; idx < 32 * 128; idx += 256) {       // theta -> [0,128)
    const int row = idx >> 7, col = idx & 127;
    Xl[row * 224 + col] = (_Float16)p.theta[(size_t)R0 * 128 + idx];
  }
  for (int idx = tid; idx < 32 * 64; idx += 256) {        // s -> [128,192)
    const int row = idx >> 6, col = idx & 63;
    Xl[row * 224 + 128 + col] = (_Float16)p.s[(size_t)R0 * 64 + idx];
  }
  {                                                        // acts -> [192,200)
    const int row = tid >> 3, col = tid & 7;               // 32*8 == 256
    Xl[row * 224 + 192 + col] = (_Float16)p.prev_acts[(size_t)R0 * 8 + tid];
  }
  for (int idx = tid; idx < 32 * 24; idx += 256) {        // flag + pad [200,224)
    const int row = idx / 24, col = idx - row * 24;
    Xl[row * 224 + 200 + col] =
        (col == 0) ? (_Float16)p.prev_flag[R0 + row] : (_Float16)0.0f;
  }
  __syncthreads();

  // ---- trunk MLP ----------------------------------------------------------
  gemm_f16_lds<224, 256, true >(Xl, Ha, p.w1, p.b1, wid, lane);  // H1
  __syncthreads();
  gemm_f16_lds<256, 256, true >(Ha, Hb, p.w2, p.b2, wid, lane);  // H2
  __syncthreads();
  gemm_f16_lds<256, 128, false>(Hb, Ha, p.w3, p.b3, wid, lane);  // N (32x128)
  __syncthreads();

  // ---- contrib = N@Wa + sampled_i*wc ; edge = N@Wb + be  (both 32x64) -----
  {
    const bool isEdge = (wid >= 4);              // wave-uniform split
    const int  ct     = wid & 3;
    const _Float16* wf = isEdge ? p.wb : p.wa;
    v8f c0 = {}, c1 = {};
#pragma unroll
    for (int kt = 0; kt < 4; ++kt) {
      v16h b  = *(const v16h*)(wf + (kt * 4 + ct) * 512 + lane * 16);
      v16h a0 = load_a_frag(Ha,            128, kt, lane);
      v16h a1 = load_a_frag(Ha + 16 * 128, 128, kt, lane);
      c0 = wmma_step(a0, b, c0);
      c1 = wmma_step(a1, b, c1);
    }
    const int n  = ct * 16 + (lane & 15);
    const int m0 = (lane >> 4) << 3;
    if (isEdge) {
      const float bn = p.be[n];
#pragma unroll
      for (int v = 0; v < 8; ++v) {
        edgeF[(m0 + v) * 64 + n]      = c0[v] + bn;
        edgeF[(16 + m0 + v) * 64 + n] = c1[v] + bn;
      }
    } else {
      const float wcn = p.wc[n];
#pragma unroll
      for (int v = 0; v < 8; ++v) {
        const int r0 = m0 + v, r1 = 16 + m0 + v;
        contribF[r0 * 64 + n] = c0[v] + p.sampled_i[R0 + r0] * wcn;
        contribF[r1 * 64 + n] = c1[v] + p.sampled_i[R0 + r1] * wcn;
      }
    }
  }
  __syncthreads();

  // ---- exclusive cumsum over the 32 agents; edgeF := excl + j*edge --------
  if (tid < 64) {
    float run = 0.0f;
    for (int nr = 0; nr < 32; ++nr) {
      const float excl = run;
      run += contribF[nr * 64 + tid];
      edgeF[nr * 64 + tid] = excl + (float)nr * edgeF[nr * 64 + tid];
    }
  }
  __syncthreads();

  // ---- per-row logit + log-prob ------------------------------------------
  if (tid < 32) {
    const int nr = tid;
    const int r  = R0 + nr;
    float acc = p.bl[0];
#pragma unroll 4
    for (int i = 0; i < 128; ++i) acc += (float)Ha[nr * 128 + i] * p.Wl[i];
#pragma unroll 4
    for (int d = 0; d < 64; ++d)  acc += edgeF[nr * 64 + d] * p.Wl[128 + d];
    acc += p.prev_flag[r] * p.Wl[192];
    const float si   = p.sampled_i[r];
    const float logp = si * log_sigmoid(acc) + (1.0f - si) * log_sigmoid(-acc);
    p.out[OUT_LOG + r] = (nr > 0) ? logp : 0.0f;
  }

  // ---- u = relu(N@Wu + bu) (32x128) --------------------------------------
  gemm_f16_lds<128, 128, true>(Ha, Hb, p.wu, p.bu, wid, lane);
  __syncthreads();

  // ---- mean = u@Wm + bm ; var = softplus(u@Wv + bv) -> global -------------
  {
    const bool isVar = (wid >= 4);               // wave-uniform split
    const int  ct    = wid & 3;
    const _Float16* wf = isVar ? p.wv : p.wm;
    const float* bias  = isVar ? p.bv : p.bm;
    v8f c0 = {}, c1 = {};
#pragma unroll
    for (int kt = 0; kt < 4; ++kt) {
      v16h b  = *(const v16h*)(wf + (kt * 4 + ct) * 512 + lane * 16);
      v16h a0 = load_a_frag(Hb,            128, kt, lane);
      v16h a1 = load_a_frag(Hb + 16 * 128, 128, kt, lane);
      c0 = wmma_step(a0, b, c0);
      c1 = wmma_step(a1, b, c1);
    }
    const int n  = ct * 16 + (lane & 15);
    const int m0 = (lane >> 4) << 3;
    const float bn = bias[n];
    float* dst = p.out + (isVar ? OUT_VAR : OUT_MEAN);
#pragma unroll
    for (int v = 0; v < 8; ++v) {
      const int r0 = R0 + m0 + v, r1 = R0 + 16 + m0 + v;
      float v0 = c0[v] + bn;
      float v1 = c1[v] + bn;
      if (isVar) { v0 = softplus_f(v0); v1 = softplus_f(v1); }
      dst[r0 * 64 + n] = v0;
      dst[r1 * 64 + n] = v1;
    }
  }
}

// ---------------------------------------------------------------------------
extern "C" void kernel_launch(void* const* d_in, const int* in_sizes, int n_in,
                              void* d_out, int out_size, void* d_ws,
                              size_t ws_size, hipStream_t stream) {
  (void)in_sizes; (void)n_in; (void)out_size; (void)ws_size;
  const float* theta     = (const float*)d_in[0];
  const float* s_in      = (const float*)d_in[1];
  const float* prev_flag = (const float*)d_in[2];
  const float* prev_acts = (const float*)d_in[3];
  const float* sampled_i = (const float*)d_in[4];
  const float* W1 = (const float*)d_in[5];
  const float* b1 = (const float*)d_in[6];
  const float* W2 = (const float*)d_in[7];
  const float* b2 = (const float*)d_in[8];
  const float* W3 = (const float*)d_in[9];
  const float* b3 = (const float*)d_in[10];
  const float* We = (const float*)d_in[11];
  const float* be = (const float*)d_in[12];
  const float* Wl = (const float*)d_in[13];
  const float* bl = (const float*)d_in[14];
  const float* Wu = (const float*)d_in[15];
  const float* bu = (const float*)d_in[16];
  const float* Wm = (const float*)d_in[17];
  const float* bm = (const float*)d_in[18];
  const float* Wv = (const float*)d_in[19];
  const float* bv = (const float*)d_in[20];

  _Float16* ws = (_Float16*)d_ws;

  // pack all weight matrices into WMMA-B fragment order (f16)
  pack_weights_kernel<<<112, 256, 0, stream>>>(W1, ws + OFF_W1, 201, 256);
  pack_weights_kernel<<<128, 256, 0, stream>>>(W2, ws + OFF_W2, 256, 256);
  pack_weights_kernel<<< 64, 256, 0, stream>>>(W3, ws + OFF_W3, 256, 128);
  pack_weights_kernel<<< 16, 256, 0, stream>>>(We,            ws + OFF_WA, 128, 64);
  pack_weights_kernel<<< 16, 256, 0, stream>>>(We + 128 * 64, ws + OFF_WB, 128, 64);
  pack_weights_kernel<<< 32, 256, 0, stream>>>(Wu, ws + OFF_WU, 128, 128);
  pack_weights_kernel<<< 16, 256, 0, stream>>>(Wm, ws + OFF_WM, 128, 64);
  pack_weights_kernel<<< 16, 256, 0, stream>>>(Wv, ws + OFF_WV, 128, 64);

  Params p;
  p.theta = theta; p.s = s_in; p.prev_flag = prev_flag;
  p.prev_acts = prev_acts; p.sampled_i = sampled_i;
  p.b1 = b1; p.b2 = b2; p.b3 = b3; p.be = be;
  p.Wl = Wl; p.bl = bl; p.bu = bu; p.bm = bm; p.bv = bv;
  p.wc = We + 2 * 128 * 64;  // We[256] row
  p.w1 = ws + OFF_W1; p.w2 = ws + OFF_W2; p.w3 = ws + OFF_W3;
  p.wa = ws + OFF_WA; p.wb = ws + OFF_WB; p.wu = ws + OFF_WU;
  p.wm = ws + OFF_WM; p.wv = ws + OFF_WV;
  p.out = (float*)d_out;

  spn_fused_kernel<<<NBP, 256, 0, stream>>>(p);
}
